// InteractLayerVec_3307124818155
// MI455X (gfx1250) — matrix operand order repaired and checked by reading, not compiled, not run on hardware
//
#include <hip/hip_runtime.h>
#include <hip/hip_bf16.h>
#include <math.h>

#define N_ATOMS   8000
#define N_PAIRS   50000
#define NF        64
#define N_DIST    20
#define K_TOT     (N_DIST * NF)        // 1280
#define HARD_CUT  5.5f
#define CUSP_REG  1e-30f
#define PAIR_TILES (N_PAIRS / 16)      // 3125 (exact)

typedef __attribute__((ext_vector_type(16))) __bf16 v16bf;
typedef __attribute__((ext_vector_type(8)))  float  v8f;
typedef int v4i __attribute__((vector_size(16)));
typedef __attribute__((address_space(1))) v4i* gas_v4i;   // global int4*
typedef __attribute__((address_space(3))) v4i* las_v4i;   // LDS int4*

// ---------------------------------------------------------------- zero accum
__global__ void zero_kernel(float* __restrict__ p, int n) {
    int i = blockIdx.x * blockDim.x + threadIdx.x;
    for (; i < n; i += gridDim.x * blockDim.x) p[i] = 0.0f;
}

// ------------------------------------------------- per-pair sensitivity/unit
__global__ void prep_pairs_kernel(const float* __restrict__ dist,
                                  const float* __restrict__ coord,
                                  const float* __restrict__ mu,
                                  const float* __restrict__ sigma,
                                  float* __restrict__ sense,   // [P, 20]
                                  float* __restrict__ unitv) { // [P, 3]
    int p = blockIdx.x * blockDim.x + threadIdx.x;
    if (p >= N_PAIRS) return;
    float d   = dist[p];
    float inv = 1.0f / d;
    float cut = 0.0f;
    if (d < HARD_CUT) {
        float c = cosf(1.5707963267948966f * d / HARD_CUT);
        cut = c * c;
    }
#pragma unroll
    for (int k = 0; k < N_DIST; ++k) {
        float nd = (inv - mu[k]) / sigma[k];
        sense[p * N_DIST + k] = expf(-0.5f * nd * nd) * cut;
    }
    unitv[p * 3 + 0] = coord[p * 3 + 0] * inv;
    unitv[p * 3 + 1] = coord[p * 3 + 1] * inv;
    unitv[p * 3 + 2] = coord[p * 3 + 2] * inv;
}

// ---------------------- W2t[o][d*64+f] = bf16(int_weights[d][o][f]), [64,1280]
__global__ void prep_weights_kernel(const float* __restrict__ iw,
                                    __bf16* __restrict__ w2t) {
    int idx = blockIdx.x * blockDim.x + threadIdx.x;
    if (idx >= NF * K_TOT) return;
    int o = idx / K_TOT;
    int k = idx % K_TOT;
    int d = k >> 6, f = k & 63;
    w2t[idx] = (__bf16)iw[(d * NF + o) * NF + f];
}

// ----------------------------------------------------------- WMMA pair GEMM
// One wave per tile of 16 pairs. acc[16 pairs x 64 out] = G_tile @ W2t^T with
// G[p, d*64+f] = sense[p,d] * feat[pair_second[p], f], built on the fly in
// bf16 fragments (f32 accumulate). Result is scattered by pair_first with
// fp32 atomics into scalar + 3 vector accumulators.
__global__ __launch_bounds__(256) void pair_gemm_kernel(
        const float* __restrict__ feat,         // [A, 64]
        const int*   __restrict__ pair_first,   // [P]
        const int*   __restrict__ pair_second,  // [P]
        const float* __restrict__ sense,        // [P, 20]
        const float* __restrict__ unitv,        // [P, 3]
        const __bf16* __restrict__ w2t,         // [64, 1280] bf16
        float* __restrict__ sacc,               // [A, 64]
        float* __restrict__ vacc) {             // [A, 3, 64]
    __shared__ float lds_feat[8][16 * NF];      // 4KB per wave, 32KB total

    const int lane = threadIdx.x & 31;
    const int wv   = threadIdx.x >> 5;
    const int tile = blockIdx.x * 8 + wv;
    const bool active = tile < PAIR_TILES;

    // warm L2/L1 with the (tiny, fully reused) bf16 weight matrix
    for (int off = threadIdx.x * 64; off < NF * K_TOT; off += 256 * 64)
        __builtin_prefetch(w2t + off, 0, 1);

    // stage the 16 gathered feature rows of this wave's tile into LDS
    float* myfeat = lds_feat[wv];
    if (active) {
#if __has_builtin(__builtin_amdgcn_global_load_async_to_lds_b128)
        // gfx1250 async copy engine: per-lane 16B global->LDS transfers,
        // tracked by ASYNCcnt (no VGPR data movement).
#pragma unroll
        for (int it = 0; it < 8; ++it) {
            int cidx = it * 32 + lane;            // 256 x float4 chunks
            int row  = cidx >> 4;
            int c4   = cidx & 15;
            int j    = pair_second[tile * 16 + row];
            __builtin_amdgcn_global_load_async_to_lds_b128(
                (gas_v4i)(feat + j * NF + c4 * 4),
                (las_v4i)(myfeat + row * NF + c4 * 4),
                0, 0);
        }
        asm volatile("s_wait_asynccnt 0x0" ::: "memory");
#else
        for (int i = lane; i < 16 * NF; i += 32) {
            int row = i >> 6, f = i & 63;
            int j = pair_second[tile * 16 + row];
            myfeat[i] = feat[j * NF + f];
        }
#endif
    }
    __syncthreads();
    if (!active) return;

    // per-lane fragment coordinates (CDNA5 16-bit A / B / f32 C layouts)
    const int prow = lane & 15;                // A-matrix row (pair in tile)
    const int hi   = lane >> 4;                // lane half
    const int p0   = tile * 16 + prow;
    const float* frow = &myfeat[prow * NF];
    const float* srow = &sense[p0 * N_DIST];

    v8f acc[4];
#pragma unroll
    for (int nt = 0; nt < 4; ++nt) acc[nt] = (v8f)0.0f;

    for (int d = 0; d < N_DIST; ++d) {         // K loop: 20 d-blocks x 2 halves
        const float s = srow[d];
#pragma unroll
        for (int h = 0; h < 2; ++h) {
            const int c = d * 2 + h;           // 32-wide K chunk index
            // A fragment: lane holds row prow, K = {kb..kb+15},
            // kb = 16*hi within the chunk; f = (h?32:0) + klocal
            const int fb = (h << 5) + (hi << 4);
            v16bf a;
#pragma unroll
            for (int e = 0; e < 8; ++e) {
                a[e]     = (__bf16)(s * frow[fb + e]);
                a[e + 8] = (__bf16)(s * frow[fb + 8 + e]);
            }
            // B fragments: lane holds column o, 16 contiguous K starting at
            // c*32 + 16*hi  -> one aligned 32B load per N-tile from W2t
            const int kb = c * 32 + (hi << 4);
#pragma unroll
            for (int nt = 0; nt < 4; ++nt) {
                const int o = nt * 16 + prow;
                v16bf b = *(const v16bf*)(w2t + o * K_TOT + kb);
                acc[nt] = __builtin_amdgcn_wmma_f32_16x16x32_bf16(
                    false, a, false, b, (short)0, acc[nt], false, false);
            }
        }
    }

    // scatter: C/D layout -> element r of lane is (m = 8*hi + r, n = nt*16+prow)
    const int mb = hi << 3;
#pragma unroll
    for (int r = 0; r < 8; ++r) {
        const int pp   = tile * 16 + mb + r;
        const int aidx = pair_first[pp];
        const float ux = unitv[pp * 3 + 0];
        const float uy = unitv[pp * 3 + 1];
        const float uz = unitv[pp * 3 + 2];
#pragma unroll
        for (int nt = 0; nt < 4; ++nt) {
            const float t = acc[nt][r];
            const int   n = nt * 16 + prow;
            atomicAdd(&sacc[aidx * NF + n], t);
            atomicAdd(&vacc[aidx * 3 * NF + n],            ux * t);
            atomicAdd(&vacc[aidx * 3 * NF + NF + n],       uy * t);
            atomicAdd(&vacc[aidx * 3 * NF + 2 * NF + n],   uz * t);
        }
    }
}

// ------------------------------------------------ norm + vecscale + self part
__global__ __launch_bounds__(64) void finalize_kernel(
        const float* __restrict__ feat,
        const float* __restrict__ self_w,    // [64, 64]
        const float* __restrict__ self_b,    // [64]
        const float* __restrict__ vecscales, // [64]
        const float* __restrict__ sacc,
        const float* __restrict__ vacc,
        float* __restrict__ out) {
    __shared__ float frow[NF];
    const int a = blockIdx.x;
    const int o = threadIdx.x;
    frow[o] = feat[a * NF + o];
    __syncthreads();
    float s = self_b[o];
    const float* wr = &self_w[o * NF];
#pragma unroll 8
    for (int f = 0; f < NF; ++f) s = fmaf(frow[f], wr[f], s);
    const float vx = vacc[a * 3 * NF + o];
    const float vy = vacc[a * 3 * NF + NF + o];
    const float vz = vacc[a * 3 * NF + 2 * NF + o];
    const float fv = sqrtf(fmaf(vx, vx, fmaf(vy, vy, fmaf(vz, vz, CUSP_REG))))
                     * vecscales[o];
    out[a * NF + o] = sacc[a * NF + o] + fv + s;
}

// ---------------------------------------------------------------- launcher
extern "C" void kernel_launch(void* const* d_in, const int* in_sizes, int n_in,
                              void* d_out, int out_size, void* d_ws, size_t ws_size,
                              hipStream_t stream) {
    const float* in_features = (const float*)d_in[0];   // [8000,64]
    const float* dist_pairs  = (const float*)d_in[1];   // [50000]
    const float* coord_pairs = (const float*)d_in[2];   // [50000,3]
    const float* int_weights = (const float*)d_in[3];   // [20,64,64]
    const float* self_w      = (const float*)d_in[4];   // [64,64]
    const float* self_b      = (const float*)d_in[5];   // [64]
    const float* vecscales   = (const float*)d_in[6];   // [64]
    const float* mu          = (const float*)d_in[7];   // [20]
    const float* sigma       = (const float*)d_in[8];   // [20]
    const int*   pair_first  = (const int*)d_in[9];     // [50000]
    const int*   pair_second = (const int*)d_in[10];    // [50000]
    float* out = (float*)d_out;

    // workspace layout (floats, then bf16 tail; all offsets 32B aligned)
    float* ws_f   = (float*)d_ws;
    float* sense  = ws_f;                                   // 1,000,000 f
    float* unitv  = sense + N_PAIRS * N_DIST;               //   150,000 f
    float* sacc   = unitv + N_PAIRS * 3;                    //   512,000 f
    float* vacc   = sacc + N_ATOMS * NF;                    // 1,536,000 f
    __bf16* w2t   = (__bf16*)(vacc + N_ATOMS * 3 * NF);     //   163,840 B

    // 1) zero the atomic accumulators (sacc & vacc are contiguous)
    {
        int n = N_ATOMS * NF + N_ATOMS * 3 * NF;
        zero_kernel<<<2048, 256, 0, stream>>>(sacc, n);
    }
    // 2) pair sensitivities + unit vectors
    prep_pairs_kernel<<<(N_PAIRS + 255) / 256, 256, 0, stream>>>(
        dist_pairs, coord_pairs, mu, sigma, sense, unitv);
    // 3) bf16 transposed weights
    prep_weights_kernel<<<(NF * K_TOT + 255) / 256, 256, 0, stream>>>(
        int_weights, w2t);
    // 4) WMMA pair GEMM + fused scatter
    {
        int nblocks = (PAIR_TILES + 7) / 8;   // 8 waves (tiles) per block
        pair_gemm_kernel<<<nblocks, 256, 0, stream>>>(
            in_features, pair_first, pair_second, sense, unitv, w2t, sacc, vacc);
    }
    // 5) finalize
    finalize_kernel<<<N_ATOMS, 64, 0, stream>>>(
        in_features, self_w, self_b, vecscales, sacc, vacc, out);
}
